// GRUD_65515431133706
// MI455X (gfx1250) — matrix operand order repaired and checked by reading, not compiled, not run on hardware
//
#include <hip/hip_runtime.h>
#include <hip/hip_bf16.h>

// GRU-D, persistent-RNN style for gfx1250 (wave32, WMMA 16x16x32 f16).
// B=1024, T=128, D=64, H=128.
//
// Key algebraic facts exploited (from the reference):
//  - stages 2..4 read the ORIGINAL X (not the previous stage output) -> independent
//  - stage 4 == stage 3 (same weights, same inputs) -> outs = Linear(stage3 outs)
//  - stage1/stage2 only contribute h_last
//
// Design: one workgroup = 16 batch rows for the full T loop. 8 waves, each owns a
// 16-wide N-slice of H. Gate weights are converted to f16 once and held in VGPRs
// across all 128 timesteps; per step the [x', h, mask] "comb" matrix is assembled
// in LDS (f16) and consumed as WMMA A-fragments; h lives in accumulator-layout
// registers.

#define BB 1024
#define TT 128
#define DD 64
#define HH 128

typedef __attribute__((ext_vector_type(16))) _Float16 v16h;
typedef __attribute__((ext_vector_type(8)))  _Float16 v8h;
typedef __attribute__((ext_vector_type(8)))  float    v8f;

__device__ __forceinline__ v8f wmma32(v16h a, v16h b, v8f c) {
  return __builtin_amdgcn_wmma_f32_16x16x32_f16(false, a, false, b, (short)0, c,
                                                false, false);
}

// B-fragment (K=32 x N=16) from row-major f32 weight W[N_total, K_total].
// Hardware layout: lane l<16 holds K={0..7,16..23} of column n=l, lane l>=16
// holds K={8..15,24..31} of column n=l-16.  Column n of B == row n of W (B = W^T).
__device__ __forceinline__ v16h load_bfrag(const float* __restrict__ W, int ldw,
                                           int n0, int k0, int lane) {
  const float* p = W + (n0 + (lane & 15)) * ldw + k0 + ((lane & 16) ? 8 : 0);
  v16h b;
#pragma unroll
  for (int i = 0; i < 16; ++i) b[i] = (_Float16)p[i < 8 ? i : i + 8];
  return b;
}

// A-fragment (M=16 x K=32) from an f16 LDS buffer with row stride ldh (halves).
// lane l<16: row m=l, K = k0+{0..7, 16..23}; lane l>=16: row m=l-16,
// K = k0+{8..15, 24..31}.
__device__ __forceinline__ v16h load_afrag(const _Float16* buf, int ldh, int k0,
                                           int lane) {
  const _Float16* p = buf + (lane & 15) * ldh + k0 + ((lane & 16) ? 8 : 0);
  v8h lo = *(const v8h*)p;
  v8h hi = *(const v8h*)(p + 16);
  return __builtin_shufflevector(lo, hi, 0, 1, 2, 3, 4, 5, 6, 7, 8, 9, 10, 11,
                                 12, 13, 14, 15);
}

__device__ __forceinline__ float sigmoidf_(float x) {
  return 1.0f / (1.0f + __expf(-x));
}

// ---------------------------------------------------------------- x_mean ----
__global__ void xmean_kernel(const float* __restrict__ X,
                             float* __restrict__ xmean) {
  int b = blockIdx.x;       // 0..B-1
  int d = threadIdx.x;      // 0..63
  const float* p = X + (b * TT) * DD + d;
  float s = 0.0f;
#pragma unroll 8
  for (int t = 0; t < TT; ++t) s += p[t * DD];
  xmean[b * DD + d] = s * (1.0f / (float)TT);
}

// ------------------------------------------------------------- recurrence ---
// STAGE==1: GRUDCell   (K = 2D+H = 256, uses mask + gamma_x + x_mean; writes h1)
// STAGE==2: GRUDCell_t (K = D+H = 192; writes h2)
// STAGE==3: GRUDCell_t + fused output head (writes outs[B,T,D])
template <int STAGE>
__global__ __launch_bounds__(256, 1) void grud_stage_kernel(
    const float* __restrict__ X, const float* __restrict__ Mask,
    const float* __restrict__ Delta, const float* __restrict__ Wz,
    const float* __restrict__ bz, const float* __restrict__ Wr,
    const float* __restrict__ br, const float* __restrict__ Wh,
    const float* __restrict__ bh, const float* __restrict__ Wgx,
    const float* __restrict__ bgx, const float* __restrict__ Wgh,
    const float* __restrict__ bgh, const float* __restrict__ Wout,
    const float* __restrict__ bout, const float* __restrict__ xmean,
    float* __restrict__ out) {
  constexpr int KCH  = (STAGE == 1) ? 8 : 6;     // K chunks of 32 for the gates
  constexpr int LDW  = (STAGE == 1) ? 256 : 192; // gate weight row stride
  constexpr int CSTR = 288;                      // comb row stride (halves)

  __shared__ _Float16 s_comb[16 * CSTR];  // [x'(0:64) | h(64:192) | mask(192:256)]
  __shared__ _Float16 s_db[16 * 72];      // delta tile, f16
  __shared__ _Float16 s_xb[16 * 72];      // x tile (stage1)
  __shared__ _Float16 s_mb[16 * 72];      // mask tile (stage1)
  __shared__ float    s_xm[16 * 64];      // x_mean tile (stage1)

  const int tid  = threadIdx.x;
  const int lane = tid & 31;
  const int w    = tid >> 5;          // wave 0..7
  const int b0   = blockIdx.x * 16;   // batch tile
  const int n0   = w * 16;            // H slice owned by this wave
  const int n    = lane & 15;         // column within tile (C/D layout)
  const int mo   = (lane >> 4) << 3;  // row offset within tile (C/D layout)

  // ---- persistent weight fragments (f16, in VGPRs for all 128 steps) ----
  v16h fr[KCH], fz[KCH], fh[KCH];
#pragma unroll
  for (int c = 0; c < KCH; ++c) {
    fr[c] = load_bfrag(Wr, LDW, n0, 32 * c, lane);
    fz[c] = load_bfrag(Wz, LDW, n0, 32 * c, lane);
    fh[c] = load_bfrag(Wh, LDW, n0, 32 * c, lane);
  }
  v16h fgh[2];
#pragma unroll
  for (int c = 0; c < 2; ++c) fgh[c] = load_bfrag(Wgh, DD, n0, 32 * c, lane);

  v16h fgx[2];
  float bgxs = 0.0f;
  if constexpr (STAGE == 1) {
    if (w < 4) {
#pragma unroll
      for (int c = 0; c < 2; ++c) fgx[c] = load_bfrag(Wgx, DD, w * 16, 32 * c, lane);
      bgxs = bgx[w * 16 + n];
    }
  }
  v16h fo[4];
  float bouts = 0.0f;
  if constexpr (STAGE == 3) {
    if (w < 4) {
#pragma unroll
      for (int c = 0; c < 4; ++c) fo[c] = load_bfrag(Wout, HH, w * 16, 32 * c, lane);
      bouts = bout[w * 16 + n];
    }
  }

  const float brs  = br[n0 + n];
  const float bzs  = bz[n0 + n];
  const float bhs  = bh[n0 + n];
  const float bghs = bgh[n0 + n];

  float hreg[8];
#pragma unroll
  for (int j = 0; j < 8; ++j) hreg[j] = 0.0f;

  if constexpr (STAGE == 1) {
    for (int i = tid; i < 16 * 64; i += 256)
      s_xm[i] = xmean[(b0 + (i >> 6)) * DD + (i & 63)];
  }
  __syncthreads();

  for (int t = 0; t < TT; ++t) {
    // (A) stage the per-step input tiles into LDS as f16
    for (int i = tid; i < 16 * 64; i += 256) {
      int m = i >> 6, d = i & 63;
      int gi = ((b0 + m) * TT + t) * DD + d;
      s_db[m * 72 + d] = (_Float16)Delta[gi];
      if constexpr (STAGE == 1) {
        float mv = Mask[gi];
        s_xb[m * 72 + d] = (_Float16)X[gi];
        s_mb[m * 72 + d] = (_Float16)mv;
        s_comb[m * CSTR + 192 + d] = (_Float16)mv;  // mask block
      } else {
        s_comb[m * CSTR + d] = (_Float16)X[gi];     // x block
      }
    }
    __syncthreads();

    // (B) gamma_h = exp(-relu(Delta @ Wgh^T + bgh)); h <- gamma_h * h
    {
      v8f acc = {0.f, 0.f, 0.f, 0.f, 0.f, 0.f, 0.f, 0.f};
      acc = wmma32(load_afrag(s_db, 72, 0, lane), fgh[0], acc);
      acc = wmma32(load_afrag(s_db, 72, 32, lane), fgh[1], acc);
#pragma unroll
      for (int j = 0; j < 8; ++j) {
        float g = __expf(-fmaxf(acc[j] + bghs, 0.0f));
        hreg[j] *= g;
        s_comb[(mo + j) * CSTR + 64 + n0 + n] = (_Float16)hreg[j];  // h block
      }
    }
    if constexpr (STAGE == 1) {
      // x' = m*x + (1-m)*(gamma_x*x + (1-gamma_x)*x_mean)
      if (w < 4) {
        v8f ax = {0.f, 0.f, 0.f, 0.f, 0.f, 0.f, 0.f, 0.f};
        ax = wmma32(load_afrag(s_db, 72, 0, lane), fgx[0], ax);
        ax = wmma32(load_afrag(s_db, 72, 32, lane), fgx[1], ax);
        int nx = w * 16 + n;
#pragma unroll
        for (int j = 0; j < 8; ++j) {
          float gx = __expf(-fmaxf(ax[j] + bgxs, 0.0f));
          int m = mo + j;
          float xv = (float)s_xb[m * 72 + nx];
          float mv = (float)s_mb[m * 72 + nx];
          float xm = s_xm[m * 64 + nx];
          float xp = mv * xv + (1.0f - mv) * (gx * xv + (1.0f - gx) * xm);
          s_comb[m * CSTR + nx] = (_Float16)xp;  // x block
        }
      }
    }
    __syncthreads();

    // (C) r, z gate pre-activations over full comb (shared A-fragments)
    float rr[8], zz[8];
    {
      v8f ar = {0.f, 0.f, 0.f, 0.f, 0.f, 0.f, 0.f, 0.f};
      v8f az = {0.f, 0.f, 0.f, 0.f, 0.f, 0.f, 0.f, 0.f};
#pragma unroll
      for (int c = 0; c < KCH; ++c) {
        v16h a = load_afrag(s_comb, CSTR, 32 * c, lane);
        ar = wmma32(a, fr[c], ar);
        az = wmma32(a, fz[c], az);
      }
#pragma unroll
      for (int j = 0; j < 8; ++j) {
        rr[j] = sigmoidf_(ar[j] + brs);
        zz[j] = sigmoidf_(az[j] + bzs);
      }
    }
    __syncthreads();  // all waves done reading comb h-block

    // (D) overwrite h block with r*h for the candidate GEMM
#pragma unroll
    for (int j = 0; j < 8; ++j)
      s_comb[(mo + j) * CSTR + 64 + n0 + n] = (_Float16)(rr[j] * hreg[j]);
    __syncthreads();

    // (E) h~ = tanh(Wh @ [x', r*h, mask] + bh); h <- (1-z)h + z*h~
    {
      v8f ah = {0.f, 0.f, 0.f, 0.f, 0.f, 0.f, 0.f, 0.f};
#pragma unroll
      for (int c = 0; c < KCH; ++c)
        ah = wmma32(load_afrag(s_comb, CSTR, 32 * c, lane), fh[c], ah);
#pragma unroll
      for (int j = 0; j < 8; ++j) {
        float hc = tanhf(ah[j] + bhs);
        hreg[j] = (1.0f - zz[j]) * hreg[j] + zz[j] * hc;
      }
    }

    if constexpr (STAGE == 3) {
      // (F) fused output head: out[t] = h_new @ Wout^T + bout
#pragma unroll
      for (int j = 0; j < 8; ++j)
        s_comb[(mo + j) * CSTR + 64 + n0 + n] = (_Float16)hreg[j];
      __syncthreads();  // also covers (E)-read vs next-(A)-write hazard
      if (w < 4) {
        v8f ao = {0.f, 0.f, 0.f, 0.f, 0.f, 0.f, 0.f, 0.f};
#pragma unroll
        for (int c = 0; c < 4; ++c)
          ao = wmma32(load_afrag(s_comb, CSTR, 64 + 32 * c, lane), fo[c], ao);
#pragma unroll
        for (int j = 0; j < 8; ++j)
          out[((b0 + mo + j) * TT + t) * DD + w * 16 + n] = ao[j] + bouts;
      }
    } else {
      __syncthreads();  // (E) reads comb x/mask blocks; next (A) rewrites them
    }
  }

  // final hidden states
  if constexpr (STAGE == 1) {
#pragma unroll
    for (int j = 0; j < 8; ++j)
      out[BB * TT * DD + (b0 + mo + j) * HH + n0 + n] = hreg[j];
  } else if constexpr (STAGE == 2) {
#pragma unroll
    for (int j = 0; j < 8; ++j)
      out[BB * TT * DD + BB * HH + (b0 + mo + j) * HH + n0 + n] = hreg[j];
  }
}

// ------------------------------------------------------------------ launch --
extern "C" void kernel_launch(void* const* d_in, const int* in_sizes, int n_in,
                              void* d_out, int out_size, void* d_ws,
                              size_t ws_size, hipStream_t stream) {
  const float* X     = (const float*)d_in[0];
  const float* Mask  = (const float*)d_in[1];
  const float* Delta = (const float*)d_in[2];
  const float* W1z = (const float*)d_in[3],  *b1z = (const float*)d_in[4];
  const float* W1r = (const float*)d_in[5],  *b1r = (const float*)d_in[6];
  const float* W1h = (const float*)d_in[7],  *b1h = (const float*)d_in[8];
  const float* W1gx = (const float*)d_in[9], *b1gx = (const float*)d_in[10];
  const float* W1gh = (const float*)d_in[11],*b1gh = (const float*)d_in[12];
  const float* W2z = (const float*)d_in[13], *b2z = (const float*)d_in[14];
  const float* W2r = (const float*)d_in[15], *b2r = (const float*)d_in[16];
  const float* W2h = (const float*)d_in[17], *b2h = (const float*)d_in[18];
  const float* W2gh = (const float*)d_in[19],*b2gh = (const float*)d_in[20];
  const float* W3z = (const float*)d_in[21], *b3z = (const float*)d_in[22];
  const float* W3r = (const float*)d_in[23], *b3r = (const float*)d_in[24];
  const float* W3h = (const float*)d_in[25], *b3h = (const float*)d_in[26];
  const float* W3gh = (const float*)d_in[27],*b3gh = (const float*)d_in[28];
  const float* Wout = (const float*)d_in[29],*bout = (const float*)d_in[30];

  float* out   = (float*)d_out;
  float* xmean = (float*)d_ws;  // B*D floats = 256 KB

  xmean_kernel<<<dim3(BB), dim3(DD), 0, stream>>>(X, xmean);

  // stage 1: needs x_mean, mask, gamma_x; only h1 (h_last) is live
  grud_stage_kernel<1><<<dim3(BB / 16), dim3(256), 0, stream>>>(
      X, Mask, Delta, W1z, b1z, W1r, b1r, W1h, b1h, W1gx, b1gx, W1gh, b1gh,
      nullptr, nullptr, xmean, out);

  // stage 2: only h2 (h_last) is live
  grud_stage_kernel<2><<<dim3(BB / 16), dim3(256), 0, stream>>>(
      X, Mask, Delta, W2z, b2z, W2r, b2r, W2h, b2h, nullptr, nullptr, W2gh,
      b2gh, nullptr, nullptr, xmean, out);

  // stage 3 (== stage 4): per-step outputs feed the fused output head
  grud_stage_kernel<3><<<dim3(BB / 16), dim3(256), 0, stream>>>(
      X, Mask, Delta, W3z, b3z, W3r, b3r, W3h, b3h, nullptr, nullptr, W3gh,
      b3gh, Wout, bout, xmean, out);
}
